// DoubleConv_12756052869398
// MI455X (gfx1250) — matrix operand order repaired and checked by reading, not compiled, not run on hardware
//
#include <hip/hip_runtime.h>

// MI455X / gfx1250: wave32, WMMA 16x16x32 bf16 -> f32 accumulate.

typedef __attribute__((ext_vector_type(16))) __bf16 v16bf;
typedef __attribute__((ext_vector_type(8)))  float  v8f;

#define DEV static __device__ __forceinline__

DEV unsigned short f2bf(float f) {           // round-to-nearest-even f32 -> bf16
  unsigned u = __float_as_uint(f);
  u += 0x7FFFu + ((u >> 16) & 1u);
  return (unsigned short)(u >> 16);
}

// ---------------------------------------------------------------------------
// Weight repack: fragment-ordered bf16 A-matrix.
// apack layout: [bp][kstep][mtile][lane][16 bf16]  (one 32B slot per lane)
// k-order: k = (r*3+s)*CIN + ci ; each 32-wide kstep stays inside one (r,s).
// A-fragment element t of lane L holds K = 16*(L>>4) + t
// (contiguous k order; the B fragment uses the same order).
// ---------------------------------------------------------------------------
template<int CIN>
__global__ void pack_weights_k(const float* __restrict__ wsh,   // [64][CIN][3][3] shared half
                               const float* __restrict__ cw,    // conv_weights [B*P][110592]
                               int cwOff,                        // 0 (layer1) or 36864 (layer2)
                               unsigned short* __restrict__ apack,
                               unsigned total) {
  unsigned idx = blockIdx.x * 256u + threadIdx.x;
  if (idx >= total) return;
  constexpr int KSTEPS = (CIN * 9) / 32;
  unsigned t    = idx & 15u;
  unsigned lane = (idx >> 4) & 31u;
  unsigned mt   = (idx >> 9) & 7u;
  unsigned rest = idx >> 12;
  unsigned ks   = rest % KSTEPS;
  unsigned bp   = rest / KSTEPS;
  unsigned h    = lane >> 4;
  unsigned vg   = t >> 1;
  unsigned kin  = 16u * h + t;
  unsigned rs   = ks / (CIN / 32);
  unsigned ci   = (ks % (CIN / 32)) * 32u + kin;
  unsigned m    = mt * 16u + (lane & 15u);
  float v;
  if (m < 64u)
    v = wsh[(m * CIN + ci) * 9u + rs];
  else
    v = cw[(size_t)bp * 110592u + (unsigned)cwOff + ((m - 64u) * CIN + ci) * 9u + rs];
  apack[idx] = f2bf(v);
}

// ---------------------------------------------------------------------------
// Fused (shared + per-patch) 3x3 conv as WMMA GEMM.
// Block = 256 thr = 8 waves; computes 128 channels x 128 pixels (4 rows of a
// 32x32 patch). Wave w owns M-tile w, all 8 N-tiles: 8 x v8f accumulators.
// Input staged once in LDS as bf16 [6 rows][34 cols][ci], stride CIP=CIN+16
// so every lane's B fragment (ci = cb..cb+15) is one aligned 32B LDS read.
// BN_IN: apply previous layer's scale/shift + ReLU while staging.
// ---------------------------------------------------------------------------
template<int CIN, bool BN_IN>
__global__ __launch_bounds__(256) void conv_gemm_k(
    const float* __restrict__ src,             // [8][CIN][256][256]
    const unsigned short* __restrict__ apack,  // fragment-ordered weights
    const float* __restrict__ scsh,            // [CIN][2] scale,shift (BN_IN only)
    float* __restrict__ dst,                   // raw conv out [8][128][256][256]
    float* __restrict__ stats)                 // [128][2] sum,sumsq (atomics)
{
  constexpr int KSTEPS = (CIN * 9) / 32;
  constexpr int CIP = CIN + 16;
  __shared__ alignas(32) unsigned short patch[6 * 34 * CIP];

  const int tid = threadIdx.x;
  const int g   = blockIdx.x & 7;        // pixel-row group (4 rows each)
  const int bp  = blockIdx.x >> 3;
  const int b   = bp >> 6;
  const int p   = bp & 63;
  const int pj  = p >> 3, pk = p & 7;
  const int Y0  = pj * 32 + g * 4 - 1;   // global row of stage row 0 (halo)
  const int X0  = pk * 32 - 1;

  for (int e = tid; e < 6 * 34 * CIN; e += 256) {
    int col = e % 34;
    int t2  = e / 34;
    int ci  = t2 % CIN;
    int row = t2 / CIN;
    int Y = Y0 + row, X = X0 + col;
    float v = 0.0f;
    if ((unsigned)Y < 256u && (unsigned)X < 256u) {
      v = src[((size_t)(b * CIN + ci) << 16) + (Y << 8) + X];
      if (BN_IN) {
        v = fmaf(v, scsh[2 * ci], scsh[2 * ci + 1]);
        v = v > 0.0f ? v : 0.0f;
      }
    }
    patch[(row * 34 + col) * CIP + ci] = f2bf(v);
  }
  __syncthreads();

  const int wave = tid >> 5;             // = M-tile
  const int lane = tid & 31;
  const int h    = lane >> 4;
  const int ln   = lane & 15;

  v8f acc[8] = {};

  const unsigned short* abase =
      apack + (size_t)bp * KSTEPS * 4096u + wave * 512 + lane * 16;

  for (int ks = 0; ks < KSTEPS; ++ks) {
    v16bf afrag = *(const v16bf*)(abase + (size_t)ks * 4096u);
    const int rs = ks / (CIN / 32);
    const int cb = (ks % (CIN / 32)) * 32 + 16 * h;
    const int r  = rs / 3;
    const int s  = rs - 3 * r;
#pragma unroll
    for (int nt = 0; nt < 8; ++nt) {
      const int row = (nt >> 1) + r;                 // output y + r (halo offset)
      const int col = ((nt & 1) << 4) + ln + s;      // output x + s
      v16bf bfrag = *(const v16bf*)&patch[(row * 34 + col) * CIP + cb];
      acc[nt] = __builtin_amdgcn_wmma_f32_16x16x32_bf16(
          false, afrag, false, bfrag, (short)0, acc[nt], false, false);
    }
  }

  // Epilogue: store raw output + per-channel BN statistics.
  const int Ybase = pj * 32 + g * 4;
  float s_[8] = {}, q_[8] = {};
#pragma unroll
  for (int nt = 0; nt < 8; ++nt) {
    const int Y = Ybase + (nt >> 1);
    const int X = pk * 32 + ((nt & 1) << 4) + ln;
#pragma unroll
    for (int r2 = 0; r2 < 8; ++r2) {
      const int ch = wave * 16 + r2 + 8 * h;         // C/D layout: M = r2 + 8*h
      float v = acc[nt][r2];
      dst[((size_t)(b * 128 + ch) << 16) + (Y << 8) + X] = v;
      s_[r2] += v;
      q_[r2] += v * v;
    }
  }
#pragma unroll
  for (int mk = 8; mk >= 1; mk >>= 1) {
#pragma unroll
    for (int r2 = 0; r2 < 8; ++r2) {
      s_[r2] += __shfl_xor(s_[r2], mk, 32);
      q_[r2] += __shfl_xor(q_[r2], mk, 32);
    }
  }
  if (ln == 0) {
#pragma unroll
    for (int r2 = 0; r2 < 8; ++r2) {
      const int ch = wave * 16 + r2 + 8 * h;
      atomicAdd(&stats[2 * ch],     s_[r2]);
      atomicAdd(&stats[2 * ch + 1], q_[r2]);
    }
  }
}

// ---------------------------------------------------------------------------
__global__ void zero_k(float* __restrict__ p, int n) {
  int i = blockIdx.x * 256 + threadIdx.x;
  if (i < n) p[i] = 0.0f;
}

__global__ void bn_stats_k(const float* __restrict__ stats,
                           const float* __restrict__ gamma,
                           const float* __restrict__ beta,
                           float* __restrict__ scsh) {
  int c = threadIdx.x;                       // 128 channels
  const float inv = 1.0f / (8.0f * 256.0f * 256.0f);
  float mean = stats[2 * c] * inv;
  float var  = stats[2 * c + 1] * inv - mean * mean;
  float sc   = gamma[c] * rsqrtf(var + 1e-5f);
  scsh[2 * c]     = sc;
  scsh[2 * c + 1] = fmaf(-mean, sc, beta[c]);
}

__global__ void finalize_k(const float* __restrict__ raw,
                           const float* __restrict__ scsh,
                           float* __restrict__ out) {
  size_t i = ((size_t)blockIdx.x * 256 + threadIdx.x) * 4;
  int ch = (int)((i >> 16) & 127);
  float sc = scsh[2 * ch], sh = scsh[2 * ch + 1];
  float4 v = *(const float4*)(raw + i);
  float4 o;
  o.x = fmaf(v.x, sc, sh); o.x = o.x > 0.0f ? o.x : 0.0f;
  o.y = fmaf(v.y, sc, sh); o.y = o.y > 0.0f ? o.y : 0.0f;
  o.z = fmaf(v.z, sc, sh); o.z = o.z > 0.0f ? o.z : 0.0f;
  o.w = fmaf(v.w, sc, sh); o.w = o.w > 0.0f ? o.w : 0.0f;
  *(float4*)(out + i) = o;
}

// ---------------------------------------------------------------------------
extern "C" void kernel_launch(void* const* d_in, const int* in_sizes, int n_in,
                              void* d_out, int out_size, void* d_ws, size_t ws_size,
                              hipStream_t stream) {
  const float* x  = (const float*)d_in[0];
  const float* cw = (const float*)d_in[1];
  const float* w1 = (const float*)d_in[2];
  const float* w2 = (const float*)d_in[3];
  const float* g1 = (const float*)d_in[4];
  const float* b1 = (const float*)d_in[5];
  const float* g2 = (const float*)d_in[6];
  const float* b2 = (const float*)d_in[7];
  float* out = (float*)d_out;

  char* ws = (char*)d_ws;
  size_t off = 0;
  float* conv1raw = (float*)(ws + off); off += (size_t)8 * 128 * 256 * 256 * 4;
  float* conv2raw = (float*)(ws + off); off += (size_t)8 * 128 * 256 * 256 * 4;
  unsigned short* apack1 = (unsigned short*)(ws + off); off += (size_t)512 * 18 * 4096 * 2;
  unsigned short* apack2 = (unsigned short*)(ws + off); off += (size_t)512 * 36 * 4096 * 2;
  float* stats1 = (float*)(ws + off); off += 256 * 4;   // [128][2]
  float* stats2 = (float*)(ws + off); off += 256 * 4;
  float* scsh1  = (float*)(ws + off); off += 256 * 4;
  float* scsh2  = (float*)(ws + off); off += 256 * 4;

  // stats1/stats2 are contiguous: zero 512 floats (graph-capture safe).
  zero_k<<<2, 256, 0, stream>>>(stats1, 512);

  unsigned tot1 = 512u * 18u * 4096u;
  pack_weights_k<64><<<(tot1 + 255) / 256, 256, 0, stream>>>(w1, cw, 0, apack1, tot1);
  unsigned tot2 = 512u * 36u * 4096u;
  pack_weights_k<128><<<(tot2 + 255) / 256, 256, 0, stream>>>(w2, cw, 36864, apack2, tot2);

  // Layer 1: conv -> raw + stats
  conv_gemm_k<64, false><<<4096, 256, 0, stream>>>(x, apack1, nullptr, conv1raw, stats1);
  bn_stats_k<<<1, 128, 0, stream>>>(stats1, g1, b1, scsh1);

  // Layer 2: BN+ReLU applied on the fly while staging conv1raw
  conv_gemm_k<128, true><<<4096, 256, 0, stream>>>(conv1raw, apack2, scsh1, conv2raw, stats2);
  bn_stats_k<<<1, 128, 0, stream>>>(stats2, g2, b2, scsh2);

  // Final BN+ReLU -> d_out
  finalize_k<<<65536, 256, 0, stream>>>(conv2raw, scsh2, out);
}
